// Self_Attention_19920058319181
// MI455X (gfx1250) — matrix operand order repaired and checked by reading, not compiled, not run on hardware
//
#include <hip/hip_runtime.h>
#include <hip/hip_bf16.h>

// Problem constants
#define Bn 8
#define Sn 2048
#define Hn 1024

typedef __attribute__((ext_vector_type(16))) __bf16 v16bf;
typedef __attribute__((ext_vector_type(8)))  __bf16 v8bf;
typedef __attribute__((ext_vector_type(8)))  float  v8f;
typedef __attribute__((ext_vector_type(4)))  unsigned int u32x4;
typedef __attribute__((ext_vector_type(8)))  int i32x8;
typedef __attribute__((ext_vector_type(4)))  int i32x4;

// Build a 16-element bf16 fragment from two 16-byte runs (VGPR0-3 = lo, VGPR4-7 = hi)
static __device__ __forceinline__ v16bf frag2(const __bf16* lo, const __bf16* hi) {
  v8bf a = *(const v8bf*)lo;
  v8bf b = *(const v8bf*)hi;
  v16bf r;
#pragma unroll
  for (int i = 0; i < 8; ++i) { r[i] = a[i]; r[i + 8] = b[i]; }
  return r;
}
// A-matrix 16x32 bf16 fragment: lanes 0-15 rows M=lane, K chunks [k0+half*8] and [+16]
static __device__ __forceinline__ v16bf frag_a(const __bf16* row, int k0, int half) {
  const __bf16* p = row + k0 + half * 8;
  return frag2(p, p + 16);
}
// B-matrix 32x16 bf16 fragment: lane = column N, K chunks [k0+half*8] and [+16]
static __device__ __forceinline__ v16bf frag_b(const __bf16* col, int k0, int half) {
  const __bf16* p = col + k0 + half * 8;
  return frag2(p, p + 16);
}
static __device__ __forceinline__ v8f wmma_bf16(v16bf a, v16bf b, v8f c) {
  return __builtin_amdgcn_wmma_f32_16x16x32_bf16(false, a, false, b, (short)0, c,
                                                 false, false);
}

// ---- Tensor Data Mover: async-load a 16-row x 1024-col bf16 tile (row stride = Hn)
// from global into LDS at byte offset lds_off. D# per CDNA5 ISA.
// 6-arg builtin form (clang-23 / therock-10.0 headers).
static __device__ __forceinline__ void tdm_load_tile_16xH(const void* gaddr,
                                                          unsigned lds_off) {
  unsigned long long ga = (unsigned long long)gaddr;
  u32x4 g0;
  g0[0] = 1u;                                        // count=1, valid user D#
  g0[1] = lds_off;                                   // lds_addr (bytes)
  g0[2] = (unsigned)(ga & 0xffffffffu);              // global_addr[31:0]
  g0[3] = (unsigned)((ga >> 32) & 0x01ffffffu) | (2u << 30);  // addr[56:32] | type=2
  i32x8 g1;
  g1[0] = 0x00010000;                 // workgroup_mask=0, data_size=1 (2 bytes)
  g1[1] = (int)(1024u << 16);         // tensor_dim0[15:0]=1024 (in [31:16])
  g1[2] = (int)(16384u << 16);        // tensor_dim0 hi=0 | tensor_dim1[15:0]=B*S
  g1[3] = (int)(1024u << 16);         // tensor_dim1 hi=0 | tile_dim0=1024
  g1[4] = 16;                         // tile_dim1=16, tile_dim2=0
  g1[5] = 1024;                       // tensor_dim0_stride=Hn (elements)
  g1[6] = 0;                          // stride0 hi, tensor_dim1_stride lo
  g1[7] = 0;                          // tensor_dim1_stride hi
  i32x4 g2 = {0, 0, 0, 0};            // tensor_dim2..: unused (2D)
  i32x4 g3 = {0, 0, 0, 0};
  i32x8 g4 = {0, 0, 0, 0, 0, 0, 0, 0};
  __builtin_amdgcn_tensor_load_to_lds(g0, g1, g2, g3, g4, 0);
}

// ---------------- Kernel 1: f32 -> bf16 convert, plus transposed copy ----------------
__global__ void cvt_x_kernel(const float* __restrict__ X, __bf16* __restrict__ Xbf,
                             __bf16* __restrict__ Xt) {
  int i = blockIdx.x * blockDim.x + threadIdx.x;
  if (i < Bn * Sn * Hn) {
    float f = X[i];
    __bf16 v = (__bf16)f;
    Xbf[i] = v;
    int b = i / (Sn * Hn);
    int rem = i % (Sn * Hn);
    int s = rem / Hn;
    int h = rem % Hn;
    Xt[((size_t)b * Hn + h) * Sn + s] = v;
  }
}

__global__ void cvt_w_kernel(const float* __restrict__ W, __bf16* __restrict__ Wbf) {
  int i = blockIdx.x * blockDim.x + threadIdx.x;
  if (i < Hn * Hn) Wbf[i] = (__bf16)W[i];
}

// ---------------- Kernel 2: QW = X @ W^T + bias  (bf16 in/out, f32 acc) -------------
// grid (B*S/16, 2), block 256 (8 waves). TDM stages the shared 16xH A tile in LDS;
// each wave computes a 16x64 strip (4 accumulators, A fragment reused 4x).
__global__ __launch_bounds__(256) void qw_gemm_kernel(
    const __bf16* __restrict__ Xbf, const __bf16* __restrict__ Wbf,
    const float* __restrict__ bias, __bf16* __restrict__ QWbf) {
  extern __shared__ char smem_raw[];
  const int wave = threadIdx.x >> 5;
  const int lane = threadIdx.x & 31;
  const int half = lane >> 4;
  const int mr   = lane & 15;

  const int row0 = blockIdx.x * 16;                       // row tile in [0, B*S)
  const int col0 = (blockIdx.y * 8 + wave) * 64;          // 64-col strip in [0, H)

  if (wave == 0) {                                        // async A-tile -> LDS[0]
    tdm_load_tile_16xH(Xbf + (size_t)row0 * Hn, 0);
    __builtin_amdgcn_s_wait_tensorcnt(0);
  }
  __syncthreads();

  const __bf16* Arow = (const __bf16*)smem_raw + mr * Hn;     // A row M = mr (LDS)
  const __bf16* B0 = Wbf + (size_t)(col0 + mr) * Hn;          // B cols N = mr + 16j
  const __bf16* B1 = B0 + (size_t)16 * Hn;
  const __bf16* B2 = B0 + (size_t)32 * Hn;
  const __bf16* B3 = B0 + (size_t)48 * Hn;

  v8f c0 = {}, c1 = {}, c2 = {}, c3 = {};
#pragma unroll 2
  for (int k0 = 0; k0 < Hn; k0 += 32) {
    __builtin_prefetch(B0 + k0 + 256, 0, 3);
    v16bf a = frag_a(Arow, k0, half);
    c0 = wmma_bf16(a, frag_b(B0, k0, half), c0);
    c1 = wmma_bf16(a, frag_b(B1, k0, half), c1);
    c2 = wmma_bf16(a, frag_b(B2, k0, half), c2);
    c3 = wmma_bf16(a, frag_b(B3, k0, half), c3);
  }
#pragma unroll
  for (int v = 0; v < 8; ++v) {
    size_t rg = (size_t)(row0 + v + 8 * half) * Hn + col0 + mr;  // M = v + 8*half
    QWbf[rg]      = (__bf16)(c0[v] + bias[col0 + mr]);
    QWbf[rg + 16] = (__bf16)(c1[v] + bias[col0 + 16 + mr]);
    QWbf[rg + 32] = (__bf16)(c2[v] + bias[col0 + 32 + mr]);
    QWbf[rg + 48] = (__bf16)(c3[v] + bias[col0 + 48 + mr]);
  }
}

// ---------------- Kernel 3: fused attention per (batch, 16-query tile) --------------
// block = 128 threads (4 waves); grid = (S/16, B)
// LDS: A-tile bf16 32KB | scores f32 [16][2048] | probs bf16 [16][2048] | reductions
#define A_OFF   0
#define SC_OFF  32768
#define PB_OFF  (SC_OFF + 16 * 2048 * 4)
#define RED_OFF (PB_OFF + 16 * 2048 * 2)
#define SMEM_BYTES (RED_OFF + (128 + 16 + 16) * 4)

__global__ __launch_bounds__(128) void attn_kernel(
    const __bf16* __restrict__ QWbf, const __bf16* __restrict__ Xbf,
    const __bf16* __restrict__ Xt, float* __restrict__ out) {
  extern __shared__ char smem_raw[];
  const __bf16* Alds = (const __bf16*)(smem_raw + A_OFF);
  float*  sc   = (float*)(smem_raw + SC_OFF);
  __bf16* pb   = (__bf16*)(smem_raw + PB_OFF);
  float*  red  = (float*)(smem_raw + RED_OFF);
  float*  mrow = red + 16 * 8;
  float*  linv = mrow + 16;

  const int tid  = threadIdx.x;
  const int wave = tid >> 5;
  const int lane = tid & 31;
  const int half = lane >> 4;
  const int mr   = lane & 15;

  const int b  = blockIdx.y;
  const int q0 = blockIdx.x * 16;

  // ---- TDM: stage QW query tile (16 x H bf16) into LDS ----
  if (wave == 0) {
    tdm_load_tile_16xH(QWbf + (size_t)(b * Sn + q0) * Hn, A_OFF);
    __builtin_amdgcn_s_wait_tensorcnt(0);
  }
  __syncthreads();

  // ---- Stage A: scores[16][2048] = QW_tile @ X^T ; 16x64 strip per wave-iter ----
  const __bf16* Arow = Alds + mr * Hn;                    // A row M = mr (LDS)
  for (int ks = wave; ks < Sn / 64; ks += 4) {
    const int k0 = ks * 64;
    const __bf16* B0 = Xbf + (size_t)(b * Sn + k0 + mr) * Hn;
    const __bf16* B1 = B0 + (size_t)16 * Hn;
    const __bf16* B2 = B0 + (size_t)32 * Hn;
    const __bf16* B3 = B0 + (size_t)48 * Hn;
    v8f c0 = {}, c1 = {}, c2 = {}, c3 = {};
#pragma unroll 2
    for (int h0 = 0; h0 < Hn; h0 += 32) {
      v16bf a = frag_a(Arow, h0, half);
      c0 = wmma_bf16(a, frag_b(B0, h0, half), c0);
      c1 = wmma_bf16(a, frag_b(B1, h0, half), c1);
      c2 = wmma_bf16(a, frag_b(B2, h0, half), c2);
      c3 = wmma_bf16(a, frag_b(B3, h0, half), c3);
    }
#pragma unroll
    for (int v = 0; v < 8; ++v) {
      float* s = sc + (v + 8 * half) * Sn + k0 + mr;
      s[0] = c0[v]; s[16] = c1[v]; s[32] = c2[v]; s[48] = c3[v];
    }
  }
  __syncthreads();

  // ---- Stage B: row max over k (16 rows x 8 segments of 256) ----
  const int r   = tid & 15;
  const int seg = tid >> 4;
  const float* rowp = sc + r * Sn + seg * 256;
  float pm = -3.0e38f;
  for (int j = 0; j < 256; ++j) pm = fmaxf(pm, rowp[j]);
  red[r * 8 + seg] = pm;
  __syncthreads();
  if (tid < 16) {
    float m = red[tid * 8];
#pragma unroll
    for (int s = 1; s < 8; ++s) m = fmaxf(m, red[tid * 8 + s]);
    mrow[tid] = m;
  }
  __syncthreads();

  // ---- Stage C: p = exp(s - m) -> bf16 probs; partial row sums ----
  {
    const float m = mrow[r];
    float ps = 0.0f;
    __bf16* prow = pb + r * Sn + seg * 256;
    for (int j = 0; j < 256; ++j) {
      float e = __expf(rowp[j] - m);
      ps += e;
      prow[j] = (__bf16)e;
    }
    red[r * 8 + seg] = ps;
  }
  __syncthreads();
  if (tid < 16) {
    float s = red[tid * 8];
#pragma unroll
    for (int k = 1; k < 8; ++k) s += red[tid * 8 + k];
    linv[tid] = 1.0f / s;
  }
  __syncthreads();

  // ---- Stage D: O = (P @ V) * 1/l ; 16x64 strip per wave-iter, A from LDS ----
  const __bf16* Ap = pb + (size_t)mr * Sn;                // A row M = mr (LDS)
  for (int hs = wave; hs < Hn / 64; hs += 4) {
    const int h0 = hs * 64;
    const __bf16* B0 = Xt + ((size_t)b * Hn + h0 + mr) * Sn;
    const __bf16* B1 = B0 + (size_t)16 * Sn;
    const __bf16* B2 = B0 + (size_t)32 * Sn;
    const __bf16* B3 = B0 + (size_t)48 * Sn;
    v8f c0 = {}, c1 = {}, c2 = {}, c3 = {};
#pragma unroll 2
    for (int k0 = 0; k0 < Sn; k0 += 32) {
      v16bf a = frag_a(Ap, k0, half);                     // ds_load_b128 x2
      c0 = wmma_bf16(a, frag_b(B0, k0, half), c0);
      c1 = wmma_bf16(a, frag_b(B1, k0, half), c1);
      c2 = wmma_bf16(a, frag_b(B2, k0, half), c2);
      c3 = wmma_bf16(a, frag_b(B3, k0, half), c3);
    }
#pragma unroll
    for (int v = 0; v < 8; ++v) {
      int q = v + 8 * half;
      float sc1 = linv[q];
      float* o = out + ((size_t)(b * Sn + q0 + q)) * Hn + h0 + mr;
      o[0] = c0[v] * sc1; o[16] = c1[v] * sc1; o[32] = c2[v] * sc1; o[48] = c3[v] * sc1;
    }
  }
}

// ---------------- Host launch ----------------
extern "C" void kernel_launch(void* const* d_in, const int* in_sizes, int n_in,
                              void* d_out, int out_size, void* d_ws, size_t ws_size,
                              hipStream_t stream) {
  const float* X    = (const float*)d_in[0];  // [B,S,H]
  const float* W    = (const float*)d_in[1];  // [H,H]
  const float* bias = (const float*)d_in[2];  // [H]
  float* out = (float*)d_out;

  char* ws = (char*)d_ws;
  const size_t xbf_bytes = (size_t)Bn * Sn * Hn * 2;  // 32 MB
  __bf16* Xbf  = (__bf16*)ws;
  __bf16* Xt   = (__bf16*)(ws + xbf_bytes);
  __bf16* Wbf  = (__bf16*)(ws + 2 * xbf_bytes);
  __bf16* QWbf = (__bf16*)(ws + 2 * xbf_bytes + (size_t)Hn * Hn * 2);

  const int NX = Bn * Sn * Hn;
  cvt_x_kernel<<<(NX + 255) / 256, 256, 0, stream>>>(X, Xbf, Xt);
  cvt_w_kernel<<<(Hn * Hn + 255) / 256, 256, 0, stream>>>(W, Wbf);

  qw_gemm_kernel<<<dim3((Bn * Sn) / 16, 2), 256, 32768, stream>>>(Xbf, Wbf, bias, QWbf);

  (void)hipFuncSetAttribute((const void*)attn_kernel,
                            hipFuncAttributeMaxDynamicSharedMemorySize, SMEM_BYTES);
  attn_kernel<<<dim3(Sn / 16, Bn), 128, SMEM_BYTES, stream>>>(QWbf, Xbf, Xt, out);
}
